// BeamSearch_30640296690452
// MI455X (gfx1250) — matrix operand order repaired and checked by reading, not compiled, not run on hardware
//
#include <hip/hip_runtime.h>
#include <hip/hip_bf16.h>
#include <math.h>

// Problem constants (fixed by the reference)
#define D       512
#define D3      1536
#define VOCAB   32000
#define NL      2
#define SRCLEN  32
#define WIDTH   4
#define MAXLEN  12

typedef __attribute__((ext_vector_type(16))) _Float16 v16h;
typedef __attribute__((ext_vector_type(8)))  _Float16 v8h;
typedef __attribute__((ext_vector_type(8)))  float    v8f;

// ---------------------------------------------------------------------------
// WMMA GEMM: C[M x (grid.x*16)] = A[M x K] * B[K x N] + bias, f32 in/out,
// f16 multiply, f32 accumulate.  One wave (32 threads) per 16x16 C tile.
// K must be a multiple of 32. Rows >= M are zero-padded (M <= 16 per tile).
// A panel is staged in LDS as f16 (uniform branches only -> no exec-masked
// scalar load chains); inner loop A fragment = two ds_load_b128 per lane.
// Fragment k order: contiguous k = 16*(lane/16) + s in both A and B (wave32).
// Dynamic LDS: 16 * K * sizeof(_Float16) bytes.
// ---------------------------------------------------------------------------
__global__ __launch_bounds__(32)
void k_wmma_gemm(const float* __restrict__ A, int lda,
                 const float* __restrict__ B, int ldb,
                 const float* __restrict__ bias,
                 float* __restrict__ C, int ldc,
                 int M, int K) {
  extern __shared__ _Float16 sA[];   // 16 rows x K, row-major
  const int lane = threadIdx.x & 31;
  const int n0   = blockIdx.x * 16;
  const int m0   = blockIdx.y * 16;
  const int col  = lane & 15;
  const bool hi  = lane >= 16;

  // ---- stage A panel (rows m0..m0+15) into LDS as f16; zero-pad rows >= M
  for (int r = 0; r < 16; ++r) {            // 'row < M' is wave-uniform
    int row = m0 + r;
    if (row < M) {
      const float* ap = A + (size_t)row * lda;
#pragma unroll 4
      for (int k = lane; k < K; k += 32) sA[r * K + k] = (_Float16)ap[k];
    } else {
#pragma unroll 4
      for (int k = lane; k < K; k += 32) sA[r * K + k] = (_Float16)0.0f;
    }
  }
  __syncthreads();

  v8f c;
  float bv = bias ? bias[n0 + col] : 0.0f;
#pragma unroll
  for (int r = 0; r < 8; ++r) c[r] = bv;

  // A fragment row for this lane (same for both lane halves)
  const _Float16* aRow = sA + col * K;
  const int aoff_lo = hi ? 16 : 0;    // K-offsets of the two contiguous runs
  const int aoff_hi = hi ? 24 : 8;

  for (int k0 = 0; k0 < K; k0 += 32) {
    // A fragment: two contiguous 8-half runs -> two ds_load_b128
    v8h alo = *(const v8h*)(aRow + k0 + aoff_lo);
    v8h ahi = *(const v8h*)(aRow + k0 + aoff_hi);
    v16h a = __builtin_shufflevector(alo, ahi, 0, 1, 2, 3, 4, 5, 6, 7,
                                     8, 9, 10, 11, 12, 13, 14, 15);
    // B fragment: lane gives column; lanes 0..15 hold K=k0..k0+15,
    //             lanes 16..31 hold K=k0+16..k0+31
    const int kb = k0 + (hi ? 16 : 0);
    const float* bp = B + (size_t)kb * ldb + n0 + col;
    v16h b;
#pragma unroll
    for (int j = 0; j < 16; ++j) b[j] = (_Float16)bp[(size_t)j * ldb];

    if (k0 + 32 < K)  // hint next B panel into cache (global_prefetch_b8)
      __builtin_prefetch(B + (size_t)(kb + 32) * ldb + n0 + col, 0, 1);

    c = __builtin_amdgcn_wmma_f32_16x16x32_f16(false, a, false, b,
                                               (short)0, c, false, false);
  }
  // C/D: lanes 0..15 rows m0..m0+7 (VGPR r), lanes 16..31 rows m0+8..m0+15
#pragma unroll
  for (int r = 0; r < 8; ++r) {
    int row = m0 + (hi ? 8 : 0) + r;
    if (row < M) C[(size_t)row * ldc + n0 + col] = c[r];
  }
}

// ---------------------------------------------------------------------------
// Embedding gather: dst[row] = emb[tokens[row]] (row length D)
// ---------------------------------------------------------------------------
__global__ void k_gather_rows(const float* __restrict__ emb,
                              const int* __restrict__ tokens,
                              float* __restrict__ dst) {
  int r = blockIdx.x, tid = threadIdx.x;
  int t = tokens[r];
  for (int d = tid; d < D; d += 256)
    dst[r * D + d] = emb[(size_t)t * D + d];
}

// ---------------------------------------------------------------------------
// Encoder recurrence for one layer. gi = X@Wih+bih precomputed (32 x 1536).
// Single workgroup, 1024 threads; h and gh staged in LDS.
// Writes per-step hidden into x (in place) and final state into h_out.
// ---------------------------------------------------------------------------
__global__ __launch_bounds__(1024)
void k_enc_recurrent(const float* __restrict__ gi,
                     const float* __restrict__ Whh,
                     const float* __restrict__ bhh,
                     float* __restrict__ x,
                     float* __restrict__ h_out) {
  __shared__ float h[D];
  __shared__ float gh[D3];
  int tid = threadIdx.x;
  for (int d = tid; d < D; d += 1024) h[d] = 0.0f;
  __syncthreads();
  for (int t = 0; t < SRCLEN; ++t) {
    for (int i = tid; i < D3; i += 1024) {
      float acc = bhh[i];
      for (int d = 0; d < D; ++d) acc += h[d] * Whh[d * D3 + i];
      gh[i] = acc;
    }
    __syncthreads();
    float hn = 0.0f;
    if (tid < D) {
      const float* git = gi + t * D3;
      float r = 1.0f / (1.0f + expf(-(git[tid]        + gh[tid])));
      float z = 1.0f / (1.0f + expf(-(git[D + tid]    + gh[D + tid])));
      float n = tanhf(git[2 * D + tid] + r * gh[2 * D + tid]);
      hn = (1.0f - z) * n + z * h[tid];
    }
    __syncthreads();
    if (tid < D) { h[tid] = hn; x[t * D + tid] = hn; }
    __syncthreads();
  }
  if (tid < D) h_out[tid] = h[tid];
}

// ---------------------------------------------------------------------------
// GRU gate combine for decoder (WIDTH beams, one layer):
// h[b][l] = GRU(gi[b], gh[b], h[b][l]);  xt[b] = new h (input to next layer)
// ---------------------------------------------------------------------------
__global__ void k_gru_combine(const float* __restrict__ gi,
                              const float* __restrict__ gh,
                              float* __restrict__ h,
                              float* __restrict__ xt, int l) {
  int b = blockIdx.x, d = threadIdx.x;
  const float* gib = gi + b * D3;
  const float* ghb = gh + b * D3;
  float hp = h[b * (NL * D) + l * D + d];
  float r = 1.0f / (1.0f + expf(-(gib[d]     + ghb[d])));
  float z = 1.0f / (1.0f + expf(-(gib[D + d] + ghb[D + d])));
  float n = tanhf(gib[2 * D + d] + r * ghb[2 * D + d]);
  float hn = (1.0f - z) * n + z * hp;
  h[b * (NL * D) + l * D + d] = hn;
  xt[b * D + d] = hn;
}

// ---------------------------------------------------------------------------
// Attention + concat: cat[b] = [top, softmax(enc_out @ top) @ enc_out]
// ---------------------------------------------------------------------------
__global__ void k_attention(const float* __restrict__ enc_out,
                            const float* __restrict__ h,
                            float* __restrict__ cat) {
  int b = blockIdx.x, tid = threadIdx.x;  // 256 threads
  __shared__ float attn[SRCLEN];
  __shared__ float smax, ssum;
  const float* top = h + b * (NL * D) + (NL - 1) * D;
  if (tid < SRCLEN) {
    float s = 0.0f;
    for (int d = 0; d < D; ++d) s += enc_out[tid * D + d] * top[d];
    attn[tid] = s;
  }
  __syncthreads();
  if (tid == 0) {
    float m = attn[0];
    for (int i = 1; i < SRCLEN; ++i) m = fmaxf(m, attn[i]);
    float su = 0.0f;
    for (int i = 0; i < SRCLEN; ++i) su += expf(attn[i] - m);
    smax = m; ssum = su;
  }
  __syncthreads();
  if (tid < SRCLEN) attn[tid] = expf(attn[tid] - smax) / ssum;
  __syncthreads();
  for (int d = tid; d < D; d += 256) {
    float c = 0.0f;
    for (int t = 0; t < SRCLEN; ++t) c += attn[t] * enc_out[t * D + d];
    cat[b * (2 * D) + d]     = top[d];
    cat[b * (2 * D) + D + d] = c;
  }
}

// ---------------------------------------------------------------------------
// log_softmax over VOCAB + top-4 (lowest-index tie-break, matches lax.top_k)
// ---------------------------------------------------------------------------
__global__ __launch_bounds__(1024)
void k_logsoftmax_top4(const float* __restrict__ logits,
                       float* __restrict__ val, int* __restrict__ idx) {
  int b = blockIdx.x, tid = threadIdx.x;
  const float* lg = logits + (size_t)b * VOCAB;
  __shared__ float sv[1024];
  __shared__ int   si[1024];
  __shared__ float s_lse;
  __shared__ int   chosen[WIDTH];
  float mx = -1e30f;
  for (int i = tid; i < VOCAB; i += 1024) mx = fmaxf(mx, lg[i]);
  sv[tid] = mx; __syncthreads();
  for (int s = 512; s > 0; s >>= 1) {
    if (tid < s) sv[tid] = fmaxf(sv[tid], sv[tid + s]);
    __syncthreads();
  }
  float gmax = sv[0]; __syncthreads();
  float sum = 0.0f;
  for (int i = tid; i < VOCAB; i += 1024) sum += expf(lg[i] - gmax);
  sv[tid] = sum; __syncthreads();
  for (int s = 512; s > 0; s >>= 1) {
    if (tid < s) sv[tid] += sv[tid + s];
    __syncthreads();
  }
  if (tid == 0) s_lse = gmax + logf(sv[0]);
  __syncthreads();
  float lse = s_lse;
  for (int pass = 0; pass < WIDTH; ++pass) {
    float bv = -1e30f; int bi = 0x7fffffff;
    for (int i = tid; i < VOCAB; i += 1024) {
      bool skip = false;
      for (int p = 0; p < pass; ++p) skip |= (chosen[p] == i);
      float v = lg[i];
      if (!skip && (v > bv || (v == bv && i < bi))) { bv = v; bi = i; }
    }
    sv[tid] = bv; si[tid] = bi; __syncthreads();
    for (int s = 512; s > 0; s >>= 1) {
      if (tid < s) {
        float ov = sv[tid + s]; int oi = si[tid + s];
        if (ov > sv[tid] || (ov == sv[tid] && oi < si[tid])) {
          sv[tid] = ov; si[tid] = oi;
        }
      }
      __syncthreads();
    }
    if (tid == 0) {
      chosen[pass]          = si[0];
      val[b * WIDTH + pass] = sv[0] - lse;
      idx[b * WIDTH + pass] = si[0];
    }
    __syncthreads();
  }
}

// ---------------------------------------------------------------------------
// First-step prep: all beams start from enc_h; first token is src[0]
// ---------------------------------------------------------------------------
__global__ __launch_bounds__(1024)
void k_prep_first(const int* __restrict__ src, const float* __restrict__ enc_h,
                  float* __restrict__ h, int* __restrict__ tok) {
  int tid = threadIdx.x;
  if (tid < WIDTH) tok[tid] = src[0];
  for (int i = tid; i < WIDTH * NL * D; i += 1024) h[i] = enc_h[i & (NL * D - 1)];
}

// ---------------------------------------------------------------------------
// Init beams after first dec_step: beam_best[:,1]=idx, prob=exp(val),
// reset h to enc_h (prefix re-decode starts with token 0, per reference).
// ---------------------------------------------------------------------------
__global__ __launch_bounds__(1024)
void k_init_beams(const float* __restrict__ val1, const int* __restrict__ idx1,
                  float* __restrict__ beam_best, float* __restrict__ prob_best,
                  float* __restrict__ h, const float* __restrict__ enc_h,
                  int* __restrict__ tok) {
  int tid = threadIdx.x;
  if (tid < WIDTH * MAXLEN) beam_best[tid] = 0.0f;
  __syncthreads();
  if (tid < WIDTH) {
    beam_best[tid * MAXLEN + 1] = (float)idx1[tid];  // beam 0's top-4
    prob_best[tid] = expf(val1[tid]);
    tok[tid] = 0;                                    // token 0 next
  }
  for (int i = tid; i < WIDTH * NL * D; i += 1024) h[i] = enc_h[i & (NL * D - 1)];
}

__global__ void k_set_tokens(const float* __restrict__ beam_best,
                             int* __restrict__ tok) {
  int tid = threadIdx.x;
  if (tid < WIDTH) tok[tid] = (int)beam_best[tid * MAXLEN + 1];
}

// ---------------------------------------------------------------------------
// Beam update: combine 16 candidates, top-4, permute hidden states by parent.
// On final round, write the winning row (12 floats) to out.
// ---------------------------------------------------------------------------
__global__ __launch_bounds__(1024)
void k_beam_update(const float* __restrict__ val1, const int* __restrict__ idx1,
                   float* __restrict__ beam_best, float* __restrict__ prob_best,
                   float* __restrict__ h, int* __restrict__ tok,
                   float* __restrict__ out, int t, int finalFlag) {
  __shared__ float h_lds[WIDTH * NL * D];
  __shared__ int   sel[WIDTH];
  int tid = threadIdx.x;
  for (int i = tid; i < WIDTH * NL * D; i += 1024) h_lds[i] = h[i];
  __syncthreads();
  if (tid == 0) {
    float po[WIDTH * WIDTH]; int ot[WIDTH * WIDTH];
    float ob[WIDTH][MAXLEN];
    for (int b = 0; b < WIDTH; ++b)
      for (int k = 0; k < MAXLEN; ++k) ob[b][k] = beam_best[b * MAXLEN + k];
    for (int j = 0; j < WIDTH * WIDTH; ++j) {
      po[j] = expf(val1[j]) * prob_best[j / WIDTH];
      ot[j] = idx1[j];
    }
    bool used[WIDTH * WIDTH];
    for (int j = 0; j < WIDTH * WIDTH; ++j) used[j] = false;
    for (int i = 0; i < WIDTH; ++i) {
      float bv = -1e30f; int bj = 0;
      for (int j = 0; j < WIDTH * WIDTH; ++j)
        if (!used[j] && po[j] > bv) { bv = po[j]; bj = j; }
      used[bj] = true; sel[i] = bj;
      int parent = bj / WIDTH;
      for (int k = 0; k < MAXLEN; ++k) beam_best[i * MAXLEN + k] = ob[parent][k];
      beam_best[i * MAXLEN + t] = (float)ot[bj];
      prob_best[i] = bv;
      tok[i] = ot[bj];
    }
    if (finalFlag)
      for (int k = 0; k < MAXLEN; ++k) out[k] = beam_best[k];  // row 0 wins
  }
  __syncthreads();
  for (int i = tid; i < WIDTH * NL * D; i += 1024) {
    int beam = i / (NL * D);
    h[i] = h_lds[(sel[beam] / WIDTH) * (NL * D) + (i % (NL * D))];
  }
}

// ---------------------------------------------------------------------------
extern "C" void kernel_launch(void* const* d_in, const int* in_sizes, int n_in,
                              void* d_out, int out_size, void* d_ws, size_t ws_size,
                              hipStream_t stream) {
  const int*   src     = (const int*)d_in[0];
  const float* enc_emb = (const float*)d_in[4];
  const float* enc_Wih = (const float*)d_in[5];
  const float* enc_Whh = (const float*)d_in[6];
  const float* enc_bih = (const float*)d_in[7];
  const float* enc_bhh = (const float*)d_in[8];
  const float* dec_emb = (const float*)d_in[9];
  const float* dec_Wih = (const float*)d_in[10];
  const float* dec_Whh = (const float*)d_in[11];
  const float* dec_bih = (const float*)d_in[12];
  const float* dec_bhh = (const float*)d_in[13];
  const float* out_W   = (const float*)d_in[14];
  const float* out_b   = (const float*)d_in[15];
  float* out = (float*)d_out;

  // workspace carve-up (floats)
  float* ws        = (float*)d_ws;
  float* enc_x     = ws;                        // 32*512   (doubles as enc_out)
  float* gi_enc    = enc_x     + SRCLEN * D;    // 32*1536
  float* enc_h     = gi_enc    + SRCLEN * D3;   // 2*512
  float* h         = enc_h     + NL * D;        // 4*2*512
  float* xt        = h         + WIDTH * NL * D;// 4*512
  float* gi        = xt        + WIDTH * D;     // 4*1536
  float* gh        = gi        + WIDTH * D3;    // 4*1536
  float* catb      = gh        + WIDTH * D3;    // 4*1024
  float* logits    = catb      + WIDTH * 2 * D; // 4*32000
  float* val1      = logits    + WIDTH * VOCAB; // 16
  float* prob_best = val1      + WIDTH * WIDTH; // 4
  float* beam_best = prob_best + WIDTH;         // 4*12
  int*   idx1      = (int*)(beam_best + WIDTH * MAXLEN); // 16
  int*   tok       = idx1 + WIDTH * WIDTH;      // 4

  const size_t shA_512  = 16 * (size_t)D * sizeof(_Float16);      // 16 KB
  const size_t shA_1024 = 16 * (size_t)(2 * D) * sizeof(_Float16); // 32 KB

  // ---- Encoder: GI = X@Wih hoisted out of the recurrence (WMMA GEMM) ----
  k_gather_rows<<<SRCLEN, 256, 0, stream>>>(enc_emb, src, enc_x);
  for (int l = 0; l < NL; ++l) {
    k_wmma_gemm<<<dim3(D3 / 16, SRCLEN / 16), 32, shA_512, stream>>>(
        enc_x, D, enc_Wih + (size_t)l * D * D3, D3,
        enc_bih + l * D3, gi_enc, D3, SRCLEN, D);
    k_enc_recurrent<<<1, 1024, 0, stream>>>(
        gi_enc, enc_Whh + (size_t)l * D * D3, enc_bhh + l * D3,
        enc_x, enc_h + l * D);
  }

  // one incremental 2-layer decoder GRU step over WIDTH beams
  auto gru_step = [&]() {
    k_gather_rows<<<WIDTH, 256, 0, stream>>>(dec_emb, tok, xt);
    for (int l = 0; l < NL; ++l) {
      k_wmma_gemm<<<dim3(D3 / 16, 1), 32, shA_512, stream>>>(
          xt, D, dec_Wih + (size_t)l * D * D3, D3,
          dec_bih + l * D3, gi, D3, WIDTH, D);
      k_wmma_gemm<<<dim3(D3 / 16, 1), 32, shA_512, stream>>>(
          h + l * D, NL * D, dec_Whh + (size_t)l * D * D3, D3,
          dec_bhh + l * D3, gh, D3, WIDTH, D);
      k_gru_combine<<<WIDTH, D, 0, stream>>>(gi, gh, h, xt, l);
    }
  };
  auto attn_logits = [&]() {
    k_attention<<<WIDTH, 256, 0, stream>>>(enc_x, h, catb);
    k_wmma_gemm<<<dim3(VOCAB / 16, 1), 32, shA_1024, stream>>>(
        catb, 2 * D, out_W, VOCAB, out_b, logits, VOCAB, WIDTH, 2 * D);
    k_logsoftmax_top4<<<WIDTH, 1024, 0, stream>>>(logits, val1, idx1);
  };

  // ---- Init: dec_step(src[0], enc_h) -> top-4 seeds the beams ----
  k_prep_first<<<1, 1024, 0, stream>>>(src, enc_h, h, tok);
  gru_step();
  attn_logits();
  k_init_beams<<<1, 1024, 0, stream>>>(val1, idx1, beam_best, prob_best, h,
                                       enc_h, tok);
  gru_step();                                    // consume token 0 (reference quirk)
  k_set_tokens<<<1, 32, 0, stream>>>(beam_best, tok);

  // ---- Beam rounds t = 2..11 (incremental; states permuted by parent) ----
  for (int t = 2; t < MAXLEN; ++t) {
    gru_step();
    attn_logits();
    k_beam_update<<<1, 1024, 0, stream>>>(val1, idx1, beam_best, prob_best, h,
                                          tok, out, t, t == MAXLEN - 1 ? 1 : 0);
  }
}
